// VideoTokenizerEMA_28338194219132
// MI455X (gfx1250) — matrix operand order repaired and checked by reading, not compiled, not run on hardware
//
#include <hip/hip_runtime.h>

// ---------------------------------------------------------------------------
// VideoTokenizerEMA forward for MI455X (gfx1250), wave32.
//   - GEMMs (~380 GFLOP, compute bound) via v_wmma_f32_16x16x32_bf16 (f32 acc)
//   - Double-buffered LDS staging via GLOBAL_LOAD_ASYNC_TO_LDS_B128 (ASYNCcnt)
//     when the toolchain exposes the builtin; sync register staging otherwise.
//   - B matrices pre-transposed to [N][K] so both A and B tiles are contiguous
//     along K (async b128-friendly); pw_w and the codebook already are [N][K].
// Sizes: B=16 T=1024 IN=1024 D=2048 NE=512 L=2 K=5, N=B*T=16384 tokens.
// ---------------------------------------------------------------------------

#define NTOK   16384
#define TDIM   1024
#define BDIM   16
#define INDIM  1024
#define DDIM   2048
#define NEMB   512
#define KW     5

#define LSTR   20            // LDS row stride in dwords: 16B-aligned, non-pow2

typedef __attribute__((ext_vector_type(16))) __bf16 v16bf;
typedef __attribute__((ext_vector_type(8)))  float  v8f;

union FragAB { unsigned int u[8]; v16bf bf; };

#if defined(__has_builtin)
#  if __has_builtin(__builtin_amdgcn_global_load_async_to_lds_b128)
#    define HAVE_ASYNC_LDS 1
#  else
#    define HAVE_ASYNC_LDS 0
#  endif
#  if __has_builtin(__builtin_amdgcn_s_wait_asynccnt)
#    define WAIT_ASYNC(n) __builtin_amdgcn_s_wait_asynccnt(n)
#  elif HAVE_ASYNC_LDS
#    define WAIT_ASYNC(n) asm volatile("s_wait_asynccnt %0" ::"i"(n) : "memory")
#  else
#    define WAIT_ASYNC(n) ((void)0)
#  endif
#else
#  define HAVE_ASYNC_LDS 0
#  define WAIT_ASYNC(n) ((void)0)
#endif

__device__ __forceinline__ void stage_b128(unsigned int* lds,
                                           const unsigned short* g) {
#if HAVE_ASYNC_LDS
    // Builtin signature (from hipcc diagnostic): AS1 int4* src, AS3 int4* dst,
    // imm offset, imm cpol.  AS1 prints as `__device__` in HIP diagnostics.
    typedef int v4i_ __attribute__((vector_size(16)));
    __builtin_amdgcn_global_load_async_to_lds_b128(
        (__attribute__((address_space(1))) v4i_*)(void*)g,
        (__attribute__((address_space(3))) v4i_*)(void*)lds, 0, 0);
#else
    *(uint4*)lds = *(const uint4*)g;   // both 16B aligned by construction
#endif
}

__device__ __forceinline__ unsigned short f2bf(float f) {
    unsigned int u = __float_as_uint(f);
    unsigned int r = u + 0x7FFFu + ((u >> 16) & 1u);   // round-to-nearest-even
    return (unsigned short)(r >> 16);
}

__device__ __forceinline__ float gelu_exact(float x) {
    return 0.5f * x * (1.0f + erff(x * 0.70710678118654752f));
}

__device__ __forceinline__ float blockReduceSum256(float v, float* sh) {
    int tid = threadIdx.x;
    sh[tid] = v; __syncthreads();
    #pragma unroll
    for (int s = 128; s > 0; s >>= 1) {
        if (tid < s) sh[tid] += sh[tid + s];
        __syncthreads();
    }
    float r = sh[0]; __syncthreads();
    return r;
}

// ---------------------------------------------------------------------------
// Tiled bf16 WMMA GEMM: C[M,N] = A[M,K] * Bt[N,K]^T + bias[N]
// 256 threads = 8 wave32s. Block tile M=256, N=64; each wave owns 2 M-tiles
// x 4 N-tiles => 8 v_wmma per K-chunk (K-chunk = 32). Double-buffered LDS,
// filled by per-lane async b128 copies (5 per thread per chunk).
// ---------------------------------------------------------------------------
__global__ __launch_bounds__(256)
void k_gemm_bf16(const unsigned short* __restrict__ Ag,   // [M][K]
                 const unsigned short* __restrict__ Btg,  // [N][K]
                 const float* __restrict__ bias,
                 float* __restrict__ C,
                 int M, int N, int K) {
    __shared__ unsigned int sA[2][256 * LSTR];   // 2 x 20 KB
    __shared__ unsigned int sB[2][64 * LSTR];    // 2 x  5 KB

    const int tid  = threadIdx.x;
    const int lane = tid & 31;
    const int wave = tid >> 5;          // 0..7
    const int half = lane >> 4;
    const int l15  = lane & 15;

    const int m0  = blockIdx.y * 256;
    const int nb0 = blockIdx.x * 64;

    v8f acc[2][4];
    #pragma unroll
    for (int mt = 0; mt < 2; ++mt)
        #pragma unroll
        for (int nt = 0; nt < 4; ++nt)
            #pragma unroll
            for (int e = 0; e < 8; ++e) acc[mt][nt][e] = 0.0f;

    // Stage one K-chunk (32 wide) into LDS buffer `buf` with b128 copies.
    auto stage = [&](int buf, int k0) {
        #pragma unroll
        for (int j = 0; j < 4; ++j) {               // A: 256 rows x 4 quads
            int p   = tid + j * 256;                // 0..1023
            int row = p >> 2;
            int q   = p & 3;
            stage_b128(&sA[buf][row * LSTR + 4 * q],
                       Ag + (size_t)(m0 + row) * K + k0 + 8 * q);
        }
        {                                           // B: 64 rows x 4 quads
            int n = tid >> 2;
            int q = tid & 3;
            stage_b128(&sB[buf][n * LSTR + 4 * q],
                       Btg + (size_t)(nb0 + n) * K + k0 + 8 * q);
        }
    };

    const int nch = K >> 5;
    stage(0, 0);
    for (int i = 0; i < nch; ++i) {
        const int cur = i & 1;
        if (i + 1 < nch) {
            stage(cur ^ 1, (i + 1) << 5);  // prefetch next chunk
            WAIT_ASYNC(5);                 // retire current chunk's 5 copies
        } else {
            WAIT_ASYNC(0);
        }
        __syncthreads();                   // all waves' fills of `cur` visible

        // A fragments: vgpr i holds K-pair
        // 8*half + i of row l15.
        FragAB a[2];
        #pragma unroll
        for (int mt = 0; mt < 2; ++mt) {
            const int arow = ((wave * 2 + mt) * 16 + l15) * LSTR;
            #pragma unroll
            for (int ii = 0; ii < 8; ++ii)
                a[mt].u[ii] = sA[cur][arow + 8 * half + ii];
        }
        // B fragments: vgpr g holds K-pair 8*half + g of column l15.
        #pragma unroll
        for (int nt = 0; nt < 4; ++nt) {
            FragAB b;
            const int bcol = (nt * 16 + l15) * LSTR + 8 * half;
            #pragma unroll
            for (int g = 0; g < 8; ++g)
                b.u[g] = sB[cur][bcol + g];
            acc[0][nt] = __builtin_amdgcn_wmma_f32_16x16x32_bf16(
                false, a[0].bf, false, b.bf, (short)0, acc[0][nt], false, false);
            acc[1][nt] = __builtin_amdgcn_wmma_f32_16x16x32_bf16(
                false, a[1].bf, false, b.bf, (short)0, acc[1][nt], false, false);
        }
        __syncthreads();                   // before buffer `cur` is re-filled
    }

    // Epilogue: D vgpr g holds row g (lanes 0-15) / g+8 (lanes 16-31).
    #pragma unroll
    for (int nt = 0; nt < 4; ++nt) {
        int col = nb0 + nt * 16 + l15;
        float bv = bias ? bias[col] : 0.0f;
        #pragma unroll
        for (int mt = 0; mt < 2; ++mt)
            #pragma unroll
            for (int g = 0; g < 8; ++g) {
                int row = m0 + (wave * 2 + mt) * 16 + g + 8 * half;
                C[(size_t)row * N + col] = acc[mt][nt][g] + bv;
            }
    }
}

// --------------------------- elementwise helpers ---------------------------

__global__ void k_cvt_bf16(const float* __restrict__ in,
                           unsigned short* __restrict__ out, long n) {
    long i = (long)blockIdx.x * blockDim.x + threadIdx.x;
    if (i < n) out[i] = f2bf(in[i]);
}

// proj_w [k=IN][n=D] -> Wt[n][k] bf16
__global__ void k_cvt_projT(const float* __restrict__ in,
                            unsigned short* __restrict__ out) {
    long i = (long)blockIdx.x * blockDim.x + threadIdx.x;  // i = n*IN + k
    if (i < (long)DDIM * INDIM) {
        int k = (int)(i & (INDIM - 1));
        int n = (int)(i >> 10);
        out[i] = f2bf(in[(size_t)k * DDIM + n]);
    }
}

// embed [e][d] -> ent[e][d] = bf16(embed[e][d] / max(||embed[e]||, eps))
__global__ __launch_bounds__(256)
void k_embed_norm(const float* __restrict__ embed,
                  unsigned short* __restrict__ ent) {
    __shared__ float sh[256];
    int e = blockIdx.x, tid = threadIdx.x;
    const float* er = embed + (size_t)e * DDIM;
    float v[8], ss = 0.0f;
    #pragma unroll
    for (int j = 0; j < 8; ++j) { v[j] = er[tid + j * 256]; ss += v[j] * v[j]; }
    float tot = blockReduceSum256(ss, sh);
    float inv = 1.0f / fmaxf(sqrtf(tot), 1e-12f);
    #pragma unroll
    for (int j = 0; j < 8; ++j)
        ent[(size_t)e * DDIM + tid + j * 256] = f2bf(v[j] * inv);
}

// proj epilogue: z = LN(gelu(x)) per row of D
__global__ __launch_bounds__(256)
void k_gelu_ln(const float* __restrict__ X, const float* __restrict__ g,
               const float* __restrict__ b, float* __restrict__ Z) {
    __shared__ float sh[256];
    int r = blockIdx.x, tid = threadIdx.x;
    const float* xr = X + (size_t)r * DDIM;
    float v[8], s = 0.0f;
    #pragma unroll
    for (int j = 0; j < 8; ++j) { v[j] = gelu_exact(xr[tid + j * 256]); s += v[j]; }
    float mean = blockReduceSum256(s, sh) * (1.0f / DDIM);
    float ss = 0.0f;
    #pragma unroll
    for (int j = 0; j < 8; ++j) { float d = v[j] - mean; ss += d * d; }
    float var = blockReduceSum256(ss, sh) * (1.0f / DDIM);
    float rstd = rsqrtf(var + 1e-5f);
    #pragma unroll
    for (int j = 0; j < 8; ++j) {
        int c = tid + j * 256;
        Z[(size_t)r * DDIM + c] = (v[j] - mean) * rstd * g[c] + b[c];
    }
}

// depthwise dilated conv along T (cross-correlation, pad = 2*dil)
__global__ void k_dwconv(const float* __restrict__ Z,
                         const float* __restrict__ dw_w,
                         const float* __restrict__ dw_b,
                         int layer, int dil, float* __restrict__ Y) {
    long i = (long)blockIdx.x * blockDim.x + threadIdx.x;
    if (i >= (long)NTOK * DDIM) return;
    int c = (int)(i & (DDIM - 1));
    long n = i >> 11;
    int t = (int)(n & (TDIM - 1));
    long bbase = (n - t) * DDIM;
    const float* w = dw_w + ((size_t)layer * DDIM + c) * KW;
    float acc = dw_b[(size_t)layer * DDIM + c];
    #pragma unroll
    for (int k = 0; k < KW; ++k) {
        int tt = t + (k - 2) * dil;
        if (tt >= 0 && tt < TDIM)
            acc += Z[bbase + (size_t)tt * DDIM + c] * w[k];
    }
    Y[i] = acc;
}

// residual epilogue: z += gelu(LN(y)) per row
__global__ __launch_bounds__(256)
void k_ln_gelu_res(const float* __restrict__ Y, const float* __restrict__ g,
                   const float* __restrict__ b, float* __restrict__ Z) {
    __shared__ float sh[256];
    int r = blockIdx.x, tid = threadIdx.x;
    const float* yr = Y + (size_t)r * DDIM;
    float v[8], s = 0.0f;
    #pragma unroll
    for (int j = 0; j < 8; ++j) { v[j] = yr[tid + j * 256]; s += v[j]; }
    float mean = blockReduceSum256(s, sh) * (1.0f / DDIM);
    float ss = 0.0f;
    #pragma unroll
    for (int j = 0; j < 8; ++j) { float d = v[j] - mean; ss += d * d; }
    float var = blockReduceSum256(ss, sh) * (1.0f / DDIM);
    float rstd = rsqrtf(var + 1e-5f);
    #pragma unroll
    for (int j = 0; j < 8; ++j) {
        int c = tid + j * 256;
        size_t o = (size_t)r * DDIM + c;
        Z[o] = Z[o] + gelu_exact((v[j] - mean) * rstd * g[c] + b[c]);
    }
}

// zn = z / max(||z||, eps) -> bf16 ; also ||zn||^2 per row
__global__ __launch_bounds__(256)
void k_znorm_bf16(const float* __restrict__ Z, unsigned short* __restrict__ Zn,
                  float* __restrict__ znsq) {
    __shared__ float sh[256];
    int r = blockIdx.x, tid = threadIdx.x;
    const float* zr = Z + (size_t)r * DDIM;
    float v[8], ss = 0.0f;
    #pragma unroll
    for (int j = 0; j < 8; ++j) { v[j] = zr[tid + j * 256]; ss += v[j] * v[j]; }
    float tot = blockReduceSum256(ss, sh);
    float inv = 1.0f / fmaxf(sqrtf(tot), 1e-12f);
    #pragma unroll
    for (int j = 0; j < 8; ++j)
        Zn[(size_t)r * DDIM + tid + j * 256] = f2bf(v[j] * inv);
    if (tid == 0) znsq[r] = tot * inv * inv;
}

__global__ void k_init_acc(float* acc) { if (threadIdx.x == 0) acc[0] = 0.0f; }

// per-token argmax of S[n, 0..511]; masked accumulate ||qn-zn||^2 = 1-2S+||zn||^2
__global__ __launch_bounds__(256)
void k_argmax(const float* __restrict__ S, const float* __restrict__ znsq,
              const int* __restrict__ lengths, int* __restrict__ idxOut,
              float* __restrict__ acc) {
    __shared__ float sv[256];
    __shared__ int   si[256];
    int n = blockIdx.x, tid = threadIdx.x;
    const float* sr = S + (size_t)n * NEMB;
    float best = -3.4e38f; int bi = 0;
    #pragma unroll
    for (int j = 0; j < 2; ++j) {
        int c = tid + j * 256;
        float s = sr[c];
        if (s > best) { best = s; bi = c; }
    }
    sv[tid] = best; si[tid] = bi; __syncthreads();
    #pragma unroll
    for (int s = 128; s > 0; s >>= 1) {
        if (tid < s) {
            if (sv[tid + s] > sv[tid] ||
                (sv[tid + s] == sv[tid] && si[tid + s] < si[tid])) {
                sv[tid] = sv[tid + s]; si[tid] = si[tid + s];
            }
        }
        __syncthreads();
    }
    if (tid == 0) {
        idxOut[n] = si[0];
        int b = n >> 10, t = n & (TDIM - 1);
        if (t < lengths[b])
            atomicAdd(acc, 1.0f - 2.0f * sv[0] + znsq[n]);
    }
}

// straight-through output: q = z + (embed[idx] - z)
__global__ void k_quant(const float* __restrict__ Z, const float* __restrict__ embed,
                        const int* __restrict__ idx, float* __restrict__ Q) {
    long i = (long)blockIdx.x * blockDim.x + threadIdx.x;
    if (i >= (long)NTOK * DDIM) return;
    long n = i >> 11;
    int d = (int)(i & (DDIM - 1));
    float zv = Z[i];
    float qv = embed[(size_t)idx[n] * DDIM + d];
    Q[i] = zv + (qv - zv);
}

__global__ void k_loss(const float* __restrict__ acc,
                       const int* __restrict__ lengths, float* __restrict__ out) {
    if (threadIdx.x == 0 && blockIdx.x == 0) {
        int nv = 0;
        for (int i = 0; i < BDIM; ++i) nv += lengths[i];
        float denom = fmaxf((float)nv, 1.0f) * (float)DDIM;
        out[0] = (nv > 0) ? 0.25f * (acc[0] / denom) : 0.0f;
    }
}

// ---------------------------------------------------------------------------
extern "C" void kernel_launch(void* const* d_in, const int* in_sizes, int n_in,
                              void* d_out, int out_size, void* d_ws, size_t ws_size,
                              hipStream_t stream) {
    (void)in_sizes; (void)n_in; (void)out_size; (void)ws_size;

    const float* x      = (const float*)d_in[0];
    const int*   lens   = (const int*)  d_in[1];
    const float* proj_w = (const float*)d_in[2];
    const float* proj_b = (const float*)d_in[3];
    const float* ln0_g  = (const float*)d_in[4];
    const float* ln0_b  = (const float*)d_in[5];
    const float* dw_w   = (const float*)d_in[6];
    const float* dw_b   = (const float*)d_in[7];
    const float* pw_w   = (const float*)d_in[8];
    const float* pw_b   = (const float*)d_in[9];
    const float* bn_g   = (const float*)d_in[10];
    const float* bn_b   = (const float*)d_in[11];
    const float* embed  = (const float*)d_in[12];

    float* outF  = (float*)d_out;
    float* outQ  = outF;                                   // [16384,2048]
    float* outZ  = outF + (size_t)NTOK * DDIM;             // [16384,2048]
    int*   outI  = (int*)(outF + 2 * (size_t)NTOK * DDIM); // [16384]
    float* outL  = outF + 2 * (size_t)NTOK * DDIM + NTOK;  // scalar

    char* w = (char*)d_ws;
    unsigned short* Abf  = (unsigned short*)(w);                    // 64 MB
    unsigned short* Wbf  = (unsigned short*)(w + (67108864));       //  8 MB
    float*          Tmp  = (float*)(w + 67108864 + 8388608);        // 128 MB
    float*          Sbuf = (float*)(w + 67108864 + 8388608 + 134217728); // 32 MB
    float*          znsq = (float*)(w + 67108864 + 8388608 + 134217728 + 33554432);
    float*          accL = znsq + NTOK;

    const long nxd  = (long)NTOK * DDIM;     // 33,554,432
    const long nxin = (long)NTOK * INDIM;    // 16,777,216

    hipLaunchKernelGGL(k_init_acc, dim3(1), dim3(1), 0, stream, accL);

    // ---- projection: z = LN(gelu(x @ proj_w + proj_b)) ----
    hipLaunchKernelGGL(k_cvt_bf16, dim3((nxin + 255) / 256), dim3(256), 0, stream,
                       x, Abf, nxin);
    hipLaunchKernelGGL(k_cvt_projT, dim3(((long)INDIM * DDIM + 255) / 256),
                       dim3(256), 0, stream, proj_w, Wbf);
    hipLaunchKernelGGL(k_gemm_bf16, dim3(DDIM / 64, NTOK / 256), dim3(256), 0,
                       stream, Abf, Wbf, proj_b, Tmp, NTOK, DDIM, INDIM);
    hipLaunchKernelGGL(k_gelu_ln, dim3(NTOK), dim3(256), 0, stream,
                       Tmp, ln0_g, ln0_b, outZ);

    // ---- temporal stack: L=2 layers ----
    for (int i = 0; i < 2; ++i) {
        int dil = 1 << i;
        hipLaunchKernelGGL(k_dwconv, dim3((nxd + 255) / 256), dim3(256), 0, stream,
                           outZ, dw_w, dw_b, i, dil, Tmp);
        hipLaunchKernelGGL(k_cvt_bf16, dim3((nxd + 255) / 256), dim3(256), 0, stream,
                           Tmp, Abf, nxd);
        // pw_w[i] is [o][c] == [N][K] already: plain convert, no transpose.
        hipLaunchKernelGGL(k_cvt_bf16, dim3(((long)DDIM * DDIM + 255) / 256),
                           dim3(256), 0, stream, pw_w + (size_t)i * DDIM * DDIM,
                           Wbf, (long)DDIM * DDIM);
        hipLaunchKernelGGL(k_gemm_bf16, dim3(DDIM / 64, NTOK / 256), dim3(256), 0,
                           stream, Abf, Wbf, pw_b + (size_t)i * DDIM, Tmp,
                           NTOK, DDIM, DDIM);
        hipLaunchKernelGGL(k_ln_gelu_res, dim3(NTOK), dim3(256), 0, stream,
                           Tmp, bn_g + (size_t)i * DDIM, bn_b + (size_t)i * DDIM,
                           outZ);
    }

    // ---- VQ: S = zn @ en^T ; argmax ; commit loss ; straight-through ----
    hipLaunchKernelGGL(k_embed_norm, dim3(NEMB), dim3(256), 0, stream, embed, Wbf);
    hipLaunchKernelGGL(k_znorm_bf16, dim3(NTOK), dim3(256), 0, stream,
                       outZ, Abf, znsq);
    hipLaunchKernelGGL(k_gemm_bf16, dim3(NEMB / 64, NTOK / 256), dim3(256), 0,
                       stream, Abf, Wbf, (const float*)nullptr, Sbuf,
                       NTOK, NEMB, DDIM);
    hipLaunchKernelGGL(k_argmax, dim3(NTOK), dim3(256), 0, stream,
                       Sbuf, znsq, lens, outI, accL);
    hipLaunchKernelGGL(k_quant, dim3((nxd + 255) / 256), dim3(256), 0, stream,
                       outZ, embed, outI, outQ);
    hipLaunchKernelGGL(k_loss, dim3(1), dim3(1), 0, stream, accL, lens, outL);
}
